// SS2DExpert_71545565216782
// MI455X (gfx1250) — matrix operand order repaired and checked by reading, not compiled, not run on hardware
//
#include <hip/hip_runtime.h>
#include <hip/hip_bf16.h>

// ---------------- problem constants ----------------
constexpr int Bb   = 16;
constexpr int Hc   = 56;
constexpr int Wc   = 56;
constexpr int DIMc = 128;
constexpr int DIc  = 128;
constexpr int Kc   = 4;     // directions
constexpr int Rc   = 4;     // dt rank
constexpr int Lc   = Hc * Wc;           // 3136
constexpr int Mc   = Bb * Lc;           // 50176 rows of GEMM
constexpr int NOUTc = 2 * DIc;          // 256 cols of in_proj output

typedef __attribute__((ext_vector_type(16))) _Float16 v16h;
typedef __attribute__((ext_vector_type(8)))  float    v8f;

#if defined(__AMDGCN__) && __has_builtin(__builtin_amdgcn_global_load_async_to_lds_b128)
#define HAVE_ASYNC_LDS 1
typedef int async_v4i __attribute__((vector_size(16)));
typedef __attribute__((address_space(1))) async_v4i* gv4i_p;   // global
typedef __attribute__((address_space(3))) async_v4i* lv4i_p;   // LDS
#else
#define HAVE_ASYNC_LDS 0
#endif

// direction-k source spatial index for scan position l
__device__ __forceinline__ int src_index(int k, int l) {
    if (k == 0) return l;
    if (k == 1) return Lc - 1 - l;
    int l2 = (k == 2) ? l : (Lc - 1 - l);
    int hi = l2 % Hc;
    int wi = l2 / Hc;
    return hi * Wc + wi;            // spatial index in row-major (h,w)
}

__device__ __forceinline__ float block_reduce_sum(float v, float* red) {
    int t = threadIdx.x;
    red[t] = v;
    __syncthreads();
    for (int s = 64; s > 0; s >>= 1) {
        if (t < s) red[t] += red[t + s];
        __syncthreads();
    }
    float r = red[0];
    __syncthreads();
    return r;
}

// ---------------- K0: zero output ----------------
__global__ void zero_kernel(float* out, int n) {
    int i = blockIdx.x * blockDim.x + threadIdx.x;
    if (i < n) out[i] = 0.f;
}

// ---------------- K1: in_proj GEMM via WMMA (f16 in, f32 acc) ----------------
// xz[M,256] = X[M,128] * in_proj_w[256,128]^T
// Block tile 128x64: 8 waves, each wave a 16x64 strip (4 accumulators sharing
// one A fragment). B panel (64 cols x 128 K = 32KB fp32) staged in LDS via
// GLOBAL_LOAD_ASYNC_TO_LDS (ASYNCcnt) when available.
__global__ __launch_bounds__(256) void gemm_inproj(const float* __restrict__ X,
                                                   const float* __restrict__ Wp,
                                                   float* __restrict__ XZ) {
    __shared__ float bpanel[64 * DIMc];          // 32 KB
    const int bn = blockIdx.x & 3;               // 4 col panels of 64
    const int bm = blockIdx.x >> 2;              // 392 row panels of 128
    const int t  = threadIdx.x;

    // ---- stage contiguous 32KB panel Wp[bn*64 .. bn*64+63][0..127] ----
    const float* src = Wp + (size_t)(bn * 64) * DIMc;
#if HAVE_ASYNC_LDS
    #pragma unroll
    for (int it = 0; it < 8; ++it) {
        int idx = it * 1024 + t * 4;             // float index, 16B per lane
        __builtin_amdgcn_global_load_async_to_lds_b128(
            (gv4i_p)(src + idx),
            (lv4i_p)(bpanel + idx),
            0, 0);
    }
#if __has_builtin(__builtin_amdgcn_s_wait_asynccnt)
    __builtin_amdgcn_s_wait_asynccnt(0);
#else
    asm volatile("s_wait_asynccnt 0x0" ::: "memory");
#endif
#else
    #pragma unroll
    for (int it = 0; it < 8; ++it) {
        int idx = it * 1024 + t * 4;
        *(float4*)(bpanel + idx) = *(const float4*)(src + idx);
    }
#endif
    __syncthreads();

    const int wave = t >> 5;
    const int lane = t & 31;
    const int m  = lane & 15;                    // row (A) / col (B) within tile
    const int hi = lane >> 4;                    // half-wave select

    const float* arow = X + (size_t)(bm * 128 + wave * 16 + m) * DIMc;

    v8f acc[4] = {};
    #pragma unroll
    for (int kk = 0; kk < DIMc; kk += 32) {
        // A 16x32 f16 layout: VGPR v<4 -> K = 2v + 8*hi ; v>=4 -> K = 8+2v + 8*hi
        v16h a;
        const int abase = kk + 8 * hi;
        #pragma unroll
        for (int v = 0; v < 8; ++v) {
            int ko = (v < 4) ? (2 * v) : (8 + 2 * v);    // 0,2,4,6,16,18,20,22
            float2 av = *(const float2*)(arow + abase + ko);
            a[2 * v]     = (_Float16)av.x;
            a[2 * v + 1] = (_Float16)av.y;
        }
        // B 32x16 f16 layout: same K order as A (VGPR j<4 -> K = 2j + 8*hi ; j>=4 -> K = 8+2j + 8*hi)
        const int bbase = kk + 8 * hi;
        #pragma unroll
        for (int tt = 0; tt < 4; ++tt) {
            v16h b;
            const float* bcol = bpanel + (tt * 16 + m) * DIMc + bbase;
            #pragma unroll
            for (int j = 0; j < 8; ++j) {
                float2 bv = *(const float2*)(bcol + ((j < 4) ? (2 * j) : (8 + 2 * j)));   // ds_load_b64
                b[2 * j]     = (_Float16)bv.x;
                b[2 * j + 1] = (_Float16)bv.y;
            }
            acc[tt] = __builtin_amdgcn_wmma_f32_16x16x32_f16(false, a, false, b,
                                                             (short)0, acc[tt],
                                                             false, false);
        }
    }
    // D layout: VGPR v -> M = v + 8*hi, N = lane&15
    #pragma unroll
    for (int tt = 0; tt < 4; ++tt) {
        #pragma unroll
        for (int v = 0; v < 8; ++v) {
            int row = bm * 128 + wave * 16 + v + 8 * hi;
            int col = bn * 64 + tt * 16 + m;
            XZ[(size_t)row * NOUTc + col] = acc[tt][v];
        }
    }
}

// ---------------- K2: depthwise 3x3 conv + bias + SiLU ----------------
// reads xi = xz[:, :, 0:128] (NHWC), writes xc as (b, l, d)
__global__ __launch_bounds__(128) void conv_kernel(const float* __restrict__ xz,
                                                   const float* __restrict__ cw,
                                                   const float* __restrict__ cb,
                                                   float* __restrict__ xc) {
    int b  = blockIdx.x / Lc;
    int l  = blockIdx.x % Lc;
    int hi = l / Wc, wi = l % Wc;
    int d  = threadIdx.x;
    float acc = cb[d];
    #pragma unroll
    for (int dy = 0; dy < 3; ++dy) {
        int h2 = hi + dy - 1;
        if ((unsigned)h2 >= (unsigned)Hc) continue;
        #pragma unroll
        for (int dx = 0; dx < 3; ++dx) {
            int w2 = wi + dx - 1;
            if ((unsigned)w2 >= (unsigned)Wc) continue;
            acc += cw[d * 9 + dy * 3 + dx] *
                   xz[((size_t)b * Lc + h2 * Wc + w2) * NOUTc + d];
        }
    }
    float s = acc / (1.f + __expf(-acc));        // SiLU
    xc[((size_t)b * Lc + l) * DIc + d] = s;
}

// ---------------- K3: x_dbl projection + dt (softplus), store dt and B,C ----------------
__global__ __launch_bounds__(128) void xdbl_dt_kernel(const float* __restrict__ xc,
                                                      const float* __restrict__ xpw,  // (K,6,DI)
                                                      const float* __restrict__ dtw,  // (K,DI,R)
                                                      const float* __restrict__ dtb,  // (K,DI)
                                                      float* __restrict__ dtbuf,      // (b,k,l,d)
                                                      float* __restrict__ bcbuf) {    // (b,k,l,2)
    __shared__ float red[128];
    __shared__ float xd[6];
    int idx = blockIdx.x;
    int l = idx % Lc; idx /= Lc;
    int k = idx % Kc;
    int b = idx / Kc;
    int d = threadIdx.x;

    int ls = src_index(k, l);
    float xsv = xc[((size_t)b * Lc + ls) * DIc + d];

    for (int cc = 0; cc < 6; ++cc) {
        float s = block_reduce_sum(xpw[(k * 6 + cc) * DIc + d] * xsv, red);
        if (d == 0) xd[cc] = s;
    }
    __syncthreads();

    float acc = dtb[k * DIc + d];
    #pragma unroll
    for (int r = 0; r < Rc; ++r) acc += dtw[(k * DIc + d) * Rc + r] * xd[r];
    float dt = (acc > 20.f) ? acc : log1pf(__expf(acc));   // softplus

    size_t base = (size_t)(b * Kc + k) * Lc + l;
    dtbuf[base * DIc + d] = dt;
    if (d == 0) { bcbuf[base * 2] = xd[4]; bcbuf[base * 2 + 1] = xd[5]; }
}

// ---------------- K4: selective scan (N=1), fused +Ds*xs ----------------
__global__ __launch_bounds__(128) void scan_kernel(const float* __restrict__ xc,
                                                   const float* __restrict__ dtbuf,
                                                   const float* __restrict__ bcbuf,
                                                   const float* __restrict__ A_log,  // (K,DI,1)
                                                   const float* __restrict__ Ds,     // (K,DI)
                                                   float* __restrict__ ybuf) {       // (b,k,l,d)
    int k = blockIdx.x % Kc;
    int b = blockIdx.x / Kc;
    int d = threadIdx.x;
    float Av = -__expf(A_log[k * DIc + d]);
    float Dv = Ds[k * DIc + d];
    float h = 0.f;
    size_t base0 = (size_t)(b * Kc + k) * Lc;
    for (int l = 0; l < Lc; ++l) {
        size_t base = base0 + l;
        // prefetch next step's dt line (gfx1250 global_prefetch path)
        if (l + 1 < Lc) __builtin_prefetch(&dtbuf[(base + 1) * DIc + d], 0, 1);
        float dt = dtbuf[base * DIc + d];
        float Bv = bcbuf[base * 2];
        float Cv = bcbuf[base * 2 + 1];
        int ls = src_index(k, l);
        float xsv = xc[((size_t)b * Lc + ls) * DIc + d];
        float dA = __expf(dt * Av);
        h = dA * h + dt * Bv * xsv;
        ybuf[base * DIc + d] = h * Cv + Dv * xsv;
    }
}

// ---------------- K5: merge 4 directions + LayerNorm + SiLU gate + spatial mean ----------------
__global__ __launch_bounds__(128) void combine_kernel(const float* __restrict__ ybuf,
                                                      const float* __restrict__ xz,
                                                      const float* __restrict__ nw,
                                                      const float* __restrict__ nb,
                                                      float* __restrict__ out) {
    __shared__ float red[128];
    int b  = blockIdx.x / Lc;
    int l  = blockIdx.x % Lc;
    int hi = l / Wc, wi = l % Wc;
    int lwh = wi * Hc + hi;
    int d  = threadIdx.x;
    size_t bk = (size_t)b * Kc;
    float v = ybuf[((bk + 0) * Lc + l) * DIc + d]
            + ybuf[((bk + 1) * Lc + (Lc - 1 - l)) * DIc + d]
            + ybuf[((bk + 2) * Lc + lwh) * DIc + d]
            + ybuf[((bk + 3) * Lc + (Lc - 1 - lwh)) * DIc + d];
    float mu  = block_reduce_sum(v, red) * (1.f / DIc);
    float dv  = v - mu;
    float var = block_reduce_sum(dv * dv, red) * (1.f / DIc);
    float yn  = dv * rsqrtf(var + 1e-5f) * nw[d] + nb[d];
    float z   = xz[((size_t)b * Lc + l) * NOUTc + DIc + d];
    float g   = z / (1.f + __expf(-z));          // SiLU gate
    atomicAdd(&out[b * DIc + d], yn * g * (1.f / Lc));
}

// ---------------- launcher ----------------
extern "C" void kernel_launch(void* const* d_in, const int* in_sizes, int n_in,
                              void* d_out, int out_size, void* d_ws, size_t ws_size,
                              hipStream_t stream) {
    const float* x         = (const float*)d_in[0];
    const float* in_proj_w = (const float*)d_in[1];
    const float* conv_w    = (const float*)d_in[2];
    const float* conv_b    = (const float*)d_in[3];
    const float* x_proj_w  = (const float*)d_in[4];
    const float* dt_proj_w = (const float*)d_in[5];
    const float* dt_proj_b = (const float*)d_in[6];
    const float* A_log     = (const float*)d_in[7];
    const float* Ds        = (const float*)d_in[8];
    const float* out_nw    = (const float*)d_in[9];
    const float* out_nb    = (const float*)d_in[10];
    float* out = (float*)d_out;

    // workspace layout (floats)
    float* ws  = (float*)d_ws;
    float* xz  = ws;                                  // M * 256
    float* xc  = xz  + (size_t)Mc * NOUTc;            // B * L * DI
    float* dtb = xc  + (size_t)Bb * Lc * DIc;         // B * K * L * DI
    float* bcb = dtb + (size_t)Bb * Kc * Lc * DIc;    // B * K * L * 2
    float* yb  = bcb + (size_t)Bb * Kc * Lc * 2;      // B * K * L * DI

    zero_kernel<<<(Bb * DIc + 255) / 256, 256, 0, stream>>>(out, Bb * DIc);

    // 392 row-panels x 4 col-panels, 256 threads (8 waves) per block
    gemm_inproj<<<(Mc / 128) * (NOUTc / 64), 256, 0, stream>>>(x, in_proj_w, xz);

    conv_kernel<<<Bb * Lc, 128, 0, stream>>>(xz, conv_w, conv_b, xc);

    xdbl_dt_kernel<<<Bb * Kc * Lc, 128, 0, stream>>>(xc, x_proj_w, dt_proj_w,
                                                     dt_proj_b, dtb, bcb);

    scan_kernel<<<Bb * Kc, 128, 0, stream>>>(xc, dtb, bcb, A_log, Ds, yb);

    combine_kernel<<<Bb * Lc, 128, 0, stream>>>(yb, xz, out_nw, out_nb, out);
}
